// GCPNetAutoEncoder_82660940579156
// MI455X (gfx1250) — matrix operand order repaired and checked by reading, not compiled, not run on hardware
//
#include <hip/hip_runtime.h>
#include <hip/hip_bf16.h>
#include <hip/hip_fp16.h>

// ---------------------------------------------------------------------------
// EGNN autoencoder for MI455X (gfx1250, wave32, WMMA).
//
// Roofline: ~84 GFLOP vs gather/scatter traffic -> memory bound. Per-layer
// working set (h16 6.4MB + agg 12.8MB + x 0.6MB) is L2-resident (192MB), so:
//  * edge MLP fully fused (E x 64 messages never touch HBM; LDS only),
//  * all GEMMs via v_wmma_f32_16x16x32_f16, f32 accumulation,
//  * h kept in a shadow f16 buffer (written in GEMM epilogues) so the random
//    per-edge row gathers move half the bytes and need no cvt in the hot loop,
//  * weights pre-swizzled into 32-lane x 16-half fragment order (32B/lane
//    loads) and staged in LDS once per block for the edge kernel,
//  * scatter-add via L2-resident global_atomic_add_f32.
// ---------------------------------------------------------------------------

#define N_NODES 50000
#define N_EDGES 800000
#define A_DIM   16
#define H_DIM   64
#define Z_DIM   32
#define INV_CNORM (1.0f / 16.0f)

typedef __attribute__((ext_vector_type(16))) _Float16 v16h;
typedef __attribute__((ext_vector_type(8)))  _Float16 v8h;
typedef __attribute__((ext_vector_type(8)))  float    v8f;

__device__ __forceinline__ v8f wmma_f16(v16h a, v16h b, v8f c) {
  // D = A(16x32 f16) * B(32x16 f16) + C(16x16 f32)
  return __builtin_amdgcn_wmma_f32_16x16x32_f16(
      /*neg_a=*/false, a, /*neg_b=*/false, b,
      /*c_mod=*/(short)0, c, /*reuse_a=*/false, /*reuse_b=*/false);
}

__device__ __forceinline__ float silu_f(float v) {
  return v * (1.0f / (1.0f + __expf(-v)));
}

__device__ __forceinline__ v16h cat16(v8h lo, v8h hi) {
  return __builtin_shufflevector(lo, hi, 0, 1, 2, 3, 4, 5, 6, 7,
                                 8, 9, 10, 11, 12, 13, 14, 15);
}

// A-fragment (16x32 f16) gathered from f16 rows (stride 64 halves). ISA
// layout: lane holds row M=lane&15; halves [0..7]=K(kb..kb+7),
// [8..15]=K(kb+16..kb+23), kb=ktile+8*(lane>>4). Two aligned 16B loads.
__device__ __forceinline__ v16h load_A_rows16(const _Float16* __restrict__ base,
                                              const int* __restrict__ rows,
                                              int ktile, int lane) {
  int r  = lane & 15;
  int kb = ktile + ((lane >> 4) << 3);
  const _Float16* rp = base + (long)rows[r] * H_DIM + kb;
  return cat16(*(const v8h*)(rp), *(const v8h*)(rp + 16));
}

// A-fragment from an LDS 16x64 f16 tile: two aligned 16B runs -> ds_load_b128.
__device__ __forceinline__ v16h load_A_lds(const _Float16* __restrict__ T,
                                           int ktile, int lane) {
  int r  = lane & 15;
  int kb = ktile + ((lane >> 4) << 3);
  const _Float16* rp = T + r * 64 + kb;
  return cat16(*(const v8h*)(rp), *(const v8h*)(rp + 16));
}

// B-fragment from fragment-ordered weights: one aligned 32B load per lane.
// Fragment f = (ktile/32)*Ntiles + (ncol0/16); slot = f*32 + lane.
__device__ __forceinline__ v16h load_B_frag(const _Float16* __restrict__ Wf,
                                            int Ntiles, int ktile, int ncol0, int lane) {
  int f = (ktile >> 5) * Ntiles + (ncol0 >> 4);
  return ((const v16h*)Wf)[(size_t)f * 32 + lane];
}

__device__ __forceinline__ v16h load_A_f32vec(const float* __restrict__ rowp, int kb) {
  v8f f0 = *(const v8f*)(rowp + kb);
  v8f f1 = *(const v8f*)(rowp + kb + 16);
  return cat16(__builtin_convertvector(f0, v8h), __builtin_convertvector(f1, v8h));
}

__device__ __forceinline__ _Float16 fetch_cat(const float* __restrict__ in0, int K0,
                                              const float* __restrict__ in1, int K1,
                                              long row, int k) {
  if (k < K0) return (_Float16)in0[row * (long)K0 + k];
  int k2 = k - K0;
  if (in1 != nullptr && k2 < K1) return (_Float16)in1[row * (long)K1 + k2];
  return (_Float16)0.0f;
}

// Concat+pad A-fragment for the node-side GEMMs. Wave-uniform fast paths:
// a 32-wide K tile entirely inside in0 or in1 -> vector loads; otherwise
// (only the zero-padded embedding tile) element-wise with padding.
__device__ __forceinline__ v16h load_A_cat(const float* __restrict__ in0, int K0,
                                           const float* __restrict__ in1, int K1,
                                           long row0, int ktile, int lane) {
  int r  = lane & 15;
  long row = row0 + r;
  int kb = (lane >> 4) << 3;
  if (ktile + 32 <= K0) {
    return load_A_f32vec(in0 + row * (long)K0, ktile + kb);
  }
  if (in1 != nullptr && ktile >= K0 && (ktile - K0) + 32 <= K1) {
    return load_A_f32vec(in1 + row * (long)K1, (ktile - K0) + kb);
  }
  v16h a;
#pragma unroll
  for (int i = 0; i < 8; ++i) {
    a[i]     = fetch_cat(in0, K0, in1, K1, row, ktile + kb + i);
    a[8 + i] = fetch_cat(in0, K0, in1, K1, row, ktile + kb + 16 + i);
  }
  return a;
}

// ---------------------------------------------------------------------------
// f32 -> f16 conversion into B-fragment order (zero K-padding).
// dst[f*512 + lane*16 + e] = W[kt*32 + (lane>>4)*8 + (e<8 ? e : 8+e)][nt*16 + (lane&15)]
// ---------------------------------------------------------------------------
__global__ void convert_w_frag_kernel(const float* __restrict__ src, _Float16* __restrict__ dst,
                                      int Krows, int Kout, int Kpad, int Ntiles) {
  int i = blockIdx.x * 256 + threadIdx.x;
  int tot = Kpad * Kout;
  if (i >= tot) return;
  int f   = i >> 9;
  int rem = i & 511;
  int l = rem >> 4, e = rem & 15;
  int kt = f / Ntiles, nt = f % Ntiles;
  int k = kt * 32 + ((l >> 4) << 3) + (e < 8 ? e : 8 + e);
  int c = nt * 16 + (l & 15);
  dst[i] = (k < Krows) ? (_Float16)src[(long)k * Kout + c] : (_Float16)0.0f;
}

__global__ void copy_f32_kernel(float* __restrict__ dst, const float* __restrict__ src, int n) {
  int i = blockIdx.x * 256 + threadIdx.x;
  if (i < n) dst[i] = src[i];
}

__global__ void add_f32_kernel(float* __restrict__ dst, const float* __restrict__ src, int n) {
  int i = blockIdx.x * 256 + threadIdx.x;
  if (i < n) dst[i] += src[i];
}

__global__ void write_x_out_kernel(float* __restrict__ out, const float* __restrict__ xw, int n3) {
  int i = blockIdx.x * 256 + threadIdx.x;
  if (i < n3) out[(long)(i / 3) * (A_DIM + 3) + A_DIM + (i % 3)] = xw[i];
}

// ---------------------------------------------------------------------------
// Fused edge kernel: one wave32 per 16-edge tile, 8 tiles per block.
//   ef = [h[src], h[dst], d2];  m = silu(silu(ef@We1+be1)@We2+be2)
//   scatter: agg[dst] += m ; xacc[dst] += rel * (m@Wx)/16
// h gathered as f16 (shadow buffer); We1/We2 staged in LDS; 24 WMMAs/tile.
// ---------------------------------------------------------------------------
__global__ __launch_bounds__(256) void egnn_edge_kernel(
    const _Float16* __restrict__ h16,      // shadow f16 h, stride 64
    const float* __restrict__ x,
    const int* __restrict__ src, const int* __restrict__ dst,
    const _Float16* __restrict__ We1f,     // 16 fragments (4 kt x 4 nt)
    const float* __restrict__ w129,        // row 128 of We1 (f32, 64)
    const float* __restrict__ be1,
    const _Float16* __restrict__ We2f,     // 8 fragments (2 kt x 4 nt)
    const float* __restrict__ be2,
    const float* __restrict__ Wx,          // 64 (f32)
    float* __restrict__ agg, float* __restrict__ xacc) {
  __shared__ __align__(32) _Float16 sW1[16 * 512];   // 16 KB
  __shared__ __align__(32) _Float16 sW2[8 * 512];    //  8 KB
  __shared__ __align__(16) _Float16 sM[8][16 * 64];  // 16 KB message tiles
  __shared__ int   sS[8][16], sD[8][16];
  __shared__ float sRel[8][16][3], sD2[8][16], sMx[8][16];
  __shared__ float sWx[64];

  int wave = threadIdx.x >> 5;
  int lane = threadIdx.x & 31;
  long tile = (long)blockIdx.x * 8 + wave;      // E/16 divisible by 8: always valid
  long e0 = tile * 16;

  // Cooperative staging: fragment-ordered 32B weight chunks + Wx -> LDS.
  {
    const v16h* g1 = (const v16h*)We1f;
    v16h* l1 = (v16h*)sW1;
    for (int i = threadIdx.x; i < 16 * 32; i += 256) l1[i] = g1[i];
    const v16h* g2 = (const v16h*)We2f;
    v16h* l2 = (v16h*)sW2;
    for (int i = threadIdx.x; i < 8 * 32; i += 256) l2[i] = g2[i];
    if (threadIdx.x < 64) sWx[threadIdx.x] = Wx[threadIdx.x];
  }
  if (lane < 16) {
    long e = e0 + lane;
    int s = src[e], d = dst[e];
    sS[wave][lane] = s; sD[wave][lane] = d;
    float rx = x[(long)s * 3 + 0] - x[(long)d * 3 + 0];
    float ry = x[(long)s * 3 + 1] - x[(long)d * 3 + 1];
    float rz = x[(long)s * 3 + 2] - x[(long)d * 3 + 2];
    sRel[wave][lane][0] = rx; sRel[wave][lane][1] = ry; sRel[wave][lane][2] = rz;
    sD2[wave][lane] = rx * rx + ry * ry + rz * rz;
  }
  __syncthreads();

  const int* rs = sS[wave];
  const int* rd = sD[wave];

  // GEMM1: [16,129] x [129,64] = h[src]@We1[0:64] + h[dst]@We1[64:128] + d2*We1[128]
#pragma unroll
  for (int nt = 0; nt < 4; ++nt) {
    v8f acc = {};
    acc = wmma_f16(load_A_rows16(h16, rs, 0,  lane), load_B_frag(sW1, 4, 0,  nt * 16, lane), acc);
    acc = wmma_f16(load_A_rows16(h16, rs, 32, lane), load_B_frag(sW1, 4, 32, nt * 16, lane), acc);
    acc = wmma_f16(load_A_rows16(h16, rd, 0,  lane), load_B_frag(sW1, 4, 64, nt * 16, lane), acc);
    acc = wmma_f16(load_A_rows16(h16, rd, 32, lane), load_B_frag(sW1, 4, 96, nt * 16, lane), acc);
    int n = nt * 16 + (lane & 15);
    float wn = w129[n], bn = be1[n];
#pragma unroll
    for (int i = 0; i < 8; ++i) {
      int M = i + ((lane >> 4) << 3);            // C/D layout: lanes 16-31 -> M+8
      float v = acc[i] + sD2[wave][M] * wn + bn; // d2 column folded in f32
      sM[wave][M * 64 + n] = (_Float16)silu_f(v);
    }
  }
  // Wave-private LDS tile: DS ops are in-order per wave, no barrier needed.

  // GEMM2: [16,64] x [64,64] -- results kept in registers, write-back after
  // all sM reads so the tile buffer can be reused in place.
  v8f acc2[4];
#pragma unroll
  for (int nt = 0; nt < 4; ++nt) {
    v8f acc = {};
    acc = wmma_f16(load_A_lds(sM[wave], 0,  lane), load_B_frag(sW2, 4, 0,  nt * 16, lane), acc);
    acc = wmma_f16(load_A_lds(sM[wave], 32, lane), load_B_frag(sW2, 4, 32, nt * 16, lane), acc);
    acc2[nt] = acc;
  }
#pragma unroll
  for (int nt = 0; nt < 4; ++nt) {
    int n = nt * 16 + (lane & 15);
    float bn = be2[n];
#pragma unroll
    for (int i = 0; i < 8; ++i) {
      int M = i + ((lane >> 4) << 3);
      sM[wave][M * 64 + n] = (_Float16)silu_f(acc2[nt][i] + bn);
    }
  }

  // mx = (m @ Wx) / C_NORM, one edge per lane; vectorized ds_load_b128 reads.
  if (lane < 16) {
    const v8h* mp = (const v8h*)(sM[wave] + lane * 64);
    float s = 0.0f;
#pragma unroll
    for (int kk = 0; kk < 8; ++kk) {
      v8h m8 = mp[kk];
#pragma unroll
      for (int j = 0; j < 8; ++j) s += (float)m8[j] * sWx[kk * 8 + j];
    }
    sMx[wave][lane] = s * INV_CNORM;
  }

  // Scatter h-messages: 16x64 f32 atomics (L2-resident); 4 b128 LDS reads/lane.
#pragma unroll
  for (int t = 0; t < 4; ++t) {
    int chunk = lane + 32 * t;          // 128 chunks of 8 halves
    int r = chunk >> 3, ck = chunk & 7;
    v8h m8 = ((const v8h*)(sM[wave] + r * 64))[ck];
    long basei = (long)sD[wave][r] * 64 + ck * 8;
#pragma unroll
    for (int j = 0; j < 8; ++j) atomicAdd(&agg[basei + j], (float)m8[j]);
  }
  // Equivariant coordinate scatter.
  if (lane < 16) {
    int d = sD[wave][lane];
    float mx = sMx[wave][lane];
    atomicAdd(&xacc[(long)d * 3 + 0], sRel[wave][lane][0] * mx);
    atomicAdd(&xacc[(long)d * 3 + 1], sRel[wave][lane][1] * mx);
    atomicAdd(&xacc[(long)d * 3 + 2], sRel[wave][lane][2] * mx);
  }
}

// ---------------------------------------------------------------------------
// Generic dense 16-row-tile GEMM: out = [res +] act( cat(in0,in1) @ W + b ).
// One wave32 per 16-row tile; K zero-padded to 32-multiples. When out16 is
// non-null (producers of h), also writes the shadow f16 copy (stride 64).
// ---------------------------------------------------------------------------
__global__ __launch_bounds__(256) void gemm16_kernel(
    const float* __restrict__ in0, int K0,
    const float* __restrict__ in1, int K1,
    const _Float16* __restrict__ Wf, int Ktiles,
    const float* __restrict__ bias,
    const float* __restrict__ res,
    float* __restrict__ out, _Float16* __restrict__ out16,
    int ldout, int Kout, int act, int Ntiles, int rowTiles) {
  int wave = threadIdx.x >> 5;
  int lane = threadIdx.x & 31;
  long tile = (long)blockIdx.x * 8 + wave;
  if (tile >= rowTiles) return;                 // wave-uniform: EXEC all-1 at WMMA
  long row0 = tile * 16;

  for (int nt = 0; nt < Ntiles; ++nt) {
    v8f acc = {};
    for (int kt = 0; kt < Ktiles; ++kt)
      acc = wmma_f16(load_A_cat(in0, K0, in1, K1, row0, kt * 32, lane),
                     load_B_frag(Wf, Ntiles, kt * 32, nt * 16, lane), acc);
    int n = nt * 16 + (lane & 15);
    float b = bias ? bias[n] : 0.0f;
#pragma unroll
    for (int i = 0; i < 8; ++i) {
      int M = i + ((lane >> 4) << 3);
      long row = row0 + M;
      float v = acc[i] + b;
      if (act) v = silu_f(v);
      if (res) v += res[row * (long)ldout + n];
      out[row * (long)ldout + n] = v;
      if (out16) out16[row * H_DIM + n] = (_Float16)v;
    }
  }
}

// ---------------------------------------------------------------------------
// Host-side orchestration (graph-capture safe: kernels + hipMemsetAsync only).
// ---------------------------------------------------------------------------
extern "C" void kernel_launch(void* const* d_in, const int* in_sizes, int n_in,
                              void* d_out, int out_size, void* d_ws, size_t ws_size,
                              hipStream_t stream) {
  (void)in_sizes; (void)n_in; (void)out_size; (void)ws_size;

  const float* h0    = (const float*)d_in[0];
  const float* x0    = (const float*)d_in[1];
  const int*   ei    = (const int*)d_in[2];
  const int*   srcI  = ei;
  const int*   dstI  = ei + N_EDGES;
  const float* W_emb = (const float*)d_in[3];
  const float* b_emb = (const float*)d_in[4];
  const float* W_lat = (const float*)d_in[5];
  const float* b_lat = (const float*)d_in[6];
  const float* W_up  = (const float*)d_in[7];
  const float* b_up  = (const float*)d_in[8];
  const float* W_out = (const float*)d_in[9];
  const float* b_out = (const float*)d_in[10];
  // Stacked per-layer params: [stack][We1,be1,We2,be2,Wx,Wn1,bn1,Wn2,bn2]
  const float* P[2][9];
  for (int s = 0; s < 2; ++s)
    for (int j = 0; j < 9; ++j) P[s][j] = (const float*)d_in[11 + s * 9 + j];

  // ---- workspace carve (~53 MB) ----
  char* base = (char*)d_ws;
  size_t off = 0;
  auto carve = [&](size_t bytes) -> void* {
    void* p = base + off;
    off = (off + bytes + 255) & ~(size_t)255;
    return p;
  };
  float*    xw   = (float*)carve((size_t)N_NODES * 3 * 4);
  float*    hbuf = (float*)carve((size_t)N_NODES * 64 * 4);
  _Float16* h16  = (_Float16*)carve((size_t)N_NODES * 64 * 2);
  float*    tbuf = (float*)carve((size_t)N_NODES * 64 * 4);
  float*    zbuf = (float*)carve((size_t)N_NODES * 32 * 4);
  float*    agg  = (float*)carve((size_t)N_NODES * 64 * 4);
  float*    xacc = (float*)carve((size_t)N_NODES * 3 * 4);
  _Float16* wh   = (_Float16*)carve((size_t)110000 * 2);

  _Float16* whEmb = wh + 0;      // 32x64   -> 4 fragments
  _Float16* whLat = wh + 2048;   // 64x32   -> 4 fragments
  _Float16* whUp  = wh + 4096;   // 32x64   -> 4 fragments
  _Float16* whOut = wh + 6144;   // 64x16   -> 2 fragments
  auto whWe1 = [&](int l) { return wh + 7168 + l * 24576; };
  auto whWe2 = [&](int l) { return wh + 7168 + l * 24576 + 8192; };
  auto whWn1 = [&](int l) { return wh + 7168 + l * 24576 + 12288; };
  auto whWn2 = [&](int l) { return wh + 7168 + l * 24576 + 20480; };

  auto cvt = [&](const float* s, _Float16* d, int Kr, int Ko, int Kp) {
    int tot = Kp * Ko;
    convert_w_frag_kernel<<<(tot + 255) / 256, 256, 0, stream>>>(s, d, Kr, Ko, Kp, Ko / 16);
  };

  // ---- f32 -> f16 fragment-ordered weight conversion (L2-resident) ----
  cvt(W_emb, whEmb, 16, 64, 32);
  cvt(W_lat, whLat, 64, 32, 64);
  cvt(W_up,  whUp,  32, 64, 32);
  cvt(W_out, whOut, 64, 16, 64);
  for (int l = 0; l < 4; ++l) {
    int st = l / 2, li = l % 2;
    cvt(P[st][0] + (size_t)li * 129 * 64, whWe1(l), 128, 64, 128);
    cvt(P[st][2] + (size_t)li * 64 * 64,  whWe2(l), 64,  64, 64);
    cvt(P[st][5] + (size_t)li * 128 * 64, whWn1(l), 128, 64, 128);
    cvt(P[st][7] + (size_t)li * 64 * 64,  whWn2(l), 64,  64, 64);
  }

  const int rowTiles  = N_NODES / 16;              // 3125 (exact)
  const int gemmBlks  = (rowTiles + 7) / 8;        // 391
  const int edgeBlks  = N_EDGES / 128;             // 6250 (exact)
  const int n3        = N_NODES * 3;

  // x working copy (inputs must not be mutated)
  copy_f32_kernel<<<(n3 + 255) / 256, 256, 0, stream>>>(xw, x0, n3);

  // Embedding: h = h0[N,16] @ W_emb[16,64] + b_emb (K zero-padded to 32)
  gemm16_kernel<<<gemmBlks, 256, 0, stream>>>(h0, A_DIM, nullptr, 0, whEmb, 1,
                                              b_emb, nullptr, hbuf, h16, 64, 64, 0, 4, rowTiles);

  for (int l = 0; l < 4; ++l) {
    int st = l / 2, li = l % 2;
    hipMemsetAsync(agg,  0, (size_t)N_NODES * 64 * 4, stream);
    hipMemsetAsync(xacc, 0, (size_t)N_NODES * 3 * 4,  stream);

    egnn_edge_kernel<<<edgeBlks, 256, 0, stream>>>(
        h16, xw, srcI, dstI,
        whWe1(l), P[st][0] + (size_t)li * 129 * 64 + 128 * 64, P[st][1] + (size_t)li * 64,
        whWe2(l), P[st][3] + (size_t)li * 64,
        P[st][4] + (size_t)li * 64, agg, xacc);

    // x += segment_sum(rel * mx) / C_NORM  (scaling folded into mx)
    add_f32_kernel<<<(n3 + 255) / 256, 256, 0, stream>>>(xw, xacc, n3);

    // Node MLP: t = silu([h, agg] @ Wn1 + bn1); h = h + t @ Wn2 + bn2
    gemm16_kernel<<<gemmBlks, 256, 0, stream>>>(hbuf, 64, agg, 64, whWn1(l), 4,
                                                P[st][6] + (size_t)li * 64, nullptr,
                                                tbuf, nullptr, 64, 64, 1, 4, rowTiles);
    gemm16_kernel<<<gemmBlks, 256, 0, stream>>>(tbuf, 64, nullptr, 0, whWn2(l), 2,
                                                P[st][8] + (size_t)li * 64, hbuf,
                                                hbuf, h16, 64, 64, 0, 4, rowTiles);

    if (l == 1) {  // encoder -> latent -> decoder input
      gemm16_kernel<<<gemmBlks, 256, 0, stream>>>(hbuf, 64, nullptr, 0, whLat, 2,
                                                  b_lat, nullptr, zbuf, nullptr, 32, 32, 0, 2, rowTiles);
      gemm16_kernel<<<gemmBlks, 256, 0, stream>>>(zbuf, 32, nullptr, 0, whUp, 1,
                                                  b_up, nullptr, hbuf, h16, 64, 64, 0, 4, rowTiles);
    }
  }

  // Output: h_rec = h @ W_out + b_out into cols [0,16); x into cols [16,19)
  gemm16_kernel<<<gemmBlks, 256, 0, stream>>>(hbuf, 64, nullptr, 0, whOut, 2,
                                              b_out, nullptr, (float*)d_out, nullptr,
                                              A_DIM + 3, A_DIM, 0, 1, rowTiles);
  write_x_out_kernel<<<(n3 + 255) / 256, 256, 0, stream>>>((float*)d_out, xw, n3);
}
